// TGNModel_7524782702608
// MI455X (gfx1250) — compile-verified, matrix operand order checked by reading, not run on hardware
//
#include <hip/hip_runtime.h>
#include <hip/hip_bf16.h>
#include <math.h>

typedef _Float16 h16v __attribute__((ext_vector_type(16)));
typedef _Float16 h8v  __attribute__((ext_vector_type(8)));
typedef float    f8v  __attribute__((ext_vector_type(8)));

#define NMEM  1000000
#define MEMD  172
#define BSZ   32768
#define KNB   10
#define NBROW (2 * BSZ * KNB)   // 655360

// ---------------------------------------------------------------------------
// WMMA fragment load: 16 rows x 32 K of f16, ISA 16-bit A/B layout.
// lane<16 : row = r0+lane,    K = {k0..k0+7, k0+16..k0+23}
// lane>=16: row = r0+lane-16, K = {k0+8..k0+15, k0+24..k0+31}
// Strides are multiples of 32 halves -> both chunks are 16B-aligned b128 loads.
// ---------------------------------------------------------------------------
__device__ __forceinline__ h16v load_frag(const _Float16* __restrict__ base, int ld,
                                          int r0, int k0, int lane) {
  int r  = lane & 15;
  int hi = lane >> 4;
  const _Float16* p = base + (size_t)(r0 + r) * (size_t)ld + (size_t)(k0 + hi * 8);
  h8v c0 = *(const h8v*)(p);
  h8v c1 = *(const h8v*)(p + 16);
  return __builtin_shufflevector(c0, c1, 0, 1, 2, 3, 4, 5, 6, 7,
                                 8, 9, 10, 11, 12, 13, 14, 15);
}

// ---------------------------------------------------------------------------
// Generic GEMM: out[M,N] = act(A[M,Kpad] @ W[Npad,Kpad]^T + bias)
// One wave per block; each wave computes one 16-row tile x NT column tiles.
// NO branches between loads and WMMAs: grid.y * NT must equal Ntiles exactly
// (weight buffers are zero-padded to Ntiles*16 rows). A-fragment is reused
// across NT back-to-back v_wmma_f32_16x16x32_f16 ops per K step.
// ---------------------------------------------------------------------------
template <int NT>
__global__ __launch_bounds__(32) void gemm_wmma_f16(
    const _Float16* __restrict__ A, int lda,
    const _Float16* __restrict__ W, int ldw,
    const float* __restrict__ bias, int Nvalid,
    float* __restrict__ outF,
    _Float16* __restrict__ outH, int ldo,
    int relu) {
  int lane = threadIdx.x;
  int m0   = blockIdx.x * 16;
  int n0   = blockIdx.y * (NT * 16);

  f8v acc[NT];
#pragma unroll
  for (int nt = 0; nt < NT; ++nt) acc[nt] = f8v{};

  for (int k0 = 0; k0 < lda; k0 += 32) {
    h16v a = load_frag(A, lda, m0, k0, lane);
#pragma unroll
    for (int nt = 0; nt < NT; ++nt) {
      h16v b = load_frag(W, ldw, n0 + nt * 16, k0, lane);
      acc[nt] = __builtin_amdgcn_wmma_f32_16x16x32_f16(
          false, a, false, b, (short)0, acc[nt], false, false);
    }
  }

  int rowbase = m0 + ((lane >> 4) << 3);
#pragma unroll
  for (int nt = 0; nt < NT; ++nt) {
    int col = n0 + nt * 16 + (lane & 15);
    float bv = (bias && col < Nvalid) ? bias[col] : 0.f;
#pragma unroll
    for (int r = 0; r < 8; ++r) {
      int row = rowbase + r;
      float v = acc[nt][r] + bv;
      if (relu) v = fmaxf(v, 0.f);
      if (outH) outH[(size_t)row * ldo + col] = (col < Nvalid) ? (_Float16)v : (_Float16)0.f;
      if (outF && col < Nvalid) outF[(size_t)row * (size_t)Nvalid + col] = v;
    }
  }
}

// ---------------------------------------------------------------------------
// Weight convert f32 [N,K] (row offset for in_W splits) -> f16 [Npad,Kpad], 0-pad
// ---------------------------------------------------------------------------
__global__ void cvt_weight(const float* __restrict__ W, int N, int K, int rowOff,
                           _Float16* __restrict__ out, int Kpad) {
  int n = blockIdx.x;
  for (int k = threadIdx.x; k < Kpad; k += blockDim.x) {
    float v = (n < N && k < K) ? W[(size_t)(rowOff + n) * K + k] : 0.f;
    out[(size_t)n * Kpad + k] = (_Float16)v;
  }
}

// x = [mem[src] | mem[dst] | sin(dt*t_w+t_b)] -> f16 (B,448); also staged
// s_src16/s_dst16 (B,192) for the GRU hidden GEMMs.
__global__ void build_x(const float* __restrict__ mem, const int* __restrict__ src,
                        const int* __restrict__ dst, const float* __restrict__ dt,
                        const float* __restrict__ t_w, const float* __restrict__ t_b,
                        _Float16* __restrict__ x16, _Float16* __restrict__ s16,
                        _Float16* __restrict__ d16) {
  int i = blockIdx.x;
  int sid = src[i], did = dst[i];
  float dtv = dt[i];
  for (int c = threadIdx.x; c < 448; c += blockDim.x) {
    float v;
    if (c < 172)      v = mem[(size_t)sid * MEMD + c];
    else if (c < 344) v = mem[(size_t)did * MEMD + (c - 172)];
    else if (c < 444) { int j = c - 344; v = sinf(dtv * t_w[j] + t_b[j]); }
    else              v = 0.f;
    x16[(size_t)i * 448 + c] = (_Float16)v;
  }
  for (int c = threadIdx.x; c < 192; c += blockDim.x) {
    _Float16 vs = (c < 172) ? (_Float16)mem[(size_t)sid * MEMD + c] : (_Float16)0.f;
    _Float16 vd = (c < 172) ? (_Float16)mem[(size_t)did * MEMD + c] : (_Float16)0.f;
    s16[(size_t)i * 192 + c] = vs;
    d16[(size_t)i * 192 + c] = vd;
  }
}

// GRU gate fusion; writes updated row (f16, stride 192, zero pad) into newmem16
__global__ void gru_kernel(const float* __restrict__ gi, const float* __restrict__ gh,
                           const float* __restrict__ mem, const int* __restrict__ ids,
                           _Float16* __restrict__ newmem16, int rowOff) {
  int i = blockIdx.x;
  int t = threadIdx.x;            // 0..191
  _Float16 out = (_Float16)0.f;
  if (t < 172) {
    size_t o = (size_t)i * 516;
    float ir = gi[o + t], iz = gi[o + 172 + t], in = gi[o + 344 + t];
    float hr = gh[o + t], hz = gh[o + 172 + t], hn = gh[o + 344 + t];
    float r = 1.f / (1.f + __expf(-(ir + hr)));
    float z = 1.f / (1.f + __expf(-(iz + hz)));
    float n = tanhf(in + r * hn);
    float h = mem[(size_t)ids[i] * MEMD + t];
    out = (_Float16)((1.f - z) * n + z * h);
  }
  newmem16[(size_t)(rowOff + i) * 192 + t] = out;
}

__global__ void fill_upd(int* __restrict__ upd, int n) {
  int i = blockIdx.x * blockDim.x + threadIdx.x;
  if (i < n) upd[i] = -1;
}
// dst pass uses base=B so atomicMax gives deterministic dst-wins semantics
__global__ void scatter_upd(int* __restrict__ upd, const int* __restrict__ ids,
                            int base, int n) {
  int i = blockIdx.x * blockDim.x + threadIdx.x;
  if (i < n) atomicMax(&upd[ids[i]], base + i);
}

// node_mem = updated_memory[concat(src,dst)] -> f16 (2B,192)
__global__ void gather_nodes(const float* __restrict__ mem,
                             const _Float16* __restrict__ newmem16,
                             const int* __restrict__ upd,
                             const int* __restrict__ src, const int* __restrict__ dst,
                             _Float16* __restrict__ node16) {
  int b = blockIdx.x;
  int id = (b < BSZ) ? src[b] : dst[b - BSZ];
  int j = upd[id];
  for (int c = threadIdx.x; c < 192; c += blockDim.x) {
    _Float16 v;
    if (j >= 0) v = newmem16[(size_t)j * 192 + c];
    else        v = (c < 172) ? (_Float16)mem[(size_t)id * MEMD + c] : (_Float16)0.f;
    node16[(size_t)b * 192 + c] = v;
  }
}

// nbr_in = [updated_memory[neigh] | tenc(neigh_dt)] -> f16 (2B*K, 288)
__global__ void gather_nbrs(const float* __restrict__ mem,
                            const _Float16* __restrict__ newmem16,
                            const int* __restrict__ upd,
                            const int* __restrict__ nids, const float* __restrict__ ndt,
                            const float* __restrict__ t_w, const float* __restrict__ t_b,
                            _Float16* __restrict__ nbr16) {
  int e = blockIdx.x;
  int id = nids[e];
  int j = upd[id];
  float dtv = ndt[e];
  const float* mrow = mem + (size_t)id * MEMD;
  __builtin_prefetch(mrow, 0, 1);
  for (int c = threadIdx.x; c < 288; c += blockDim.x) {
    float v;
    if (c < 172)      v = (j >= 0) ? (float)newmem16[(size_t)j * 192 + c] : mrow[c];
    else if (c < 272) { int q = c - 172; v = sinf(dtv * t_w[q] + t_b[q]); }
    else              v = 0.f;
    nbr16[(size_t)e * 288 + c] = (_Float16)v;
  }
}

// 2-head attention over K=10 neighbors, hd=50; outputs ao f16 (2B,128 pad)
__global__ void attn_kernel(const _Float16* __restrict__ qh, const _Float16* __restrict__ kh,
                            const _Float16* __restrict__ vh, _Float16* __restrict__ ao) {
  __shared__ float logit[20];
  __shared__ float prob[20];
  int b = blockIdx.x;
  int t = threadIdx.x;            // 128 threads
  if (t < 20) {
    int h = t / 10, k = t % 10;
    const _Float16* qp = qh + (size_t)b * 128 + h * 50;
    const _Float16* kp = kh + ((size_t)b * KNB + k) * 128 + h * 50;
    float s = 0.f;
    for (int d = 0; d < 50; ++d) s += (float)qp[d] * (float)kp[d];
    logit[t] = s * 0.14142135623f;   // 1/sqrt(50)
  }
  __syncthreads();
  if (t < 2) {
    float mx = -1e30f;
    for (int k = 0; k < 10; ++k) mx = fmaxf(mx, logit[t * 10 + k]);
    float sum = 0.f;
    for (int k = 0; k < 10; ++k) { float e = __expf(logit[t * 10 + k] - mx); prob[t * 10 + k] = e; sum += e; }
    float inv = 1.f / sum;
    for (int k = 0; k < 10; ++k) prob[t * 10 + k] *= inv;
  }
  __syncthreads();
  _Float16 v = (_Float16)0.f;
  if (t < 100) {
    int h = t / 50;
    float s = 0.f;
    for (int k = 0; k < 10; ++k)
      s += prob[h * 10 + k] * (float)vh[((size_t)b * KNB + k) * 128 + t];
    v = (_Float16)s;
  }
  ao[(size_t)b * 128 + t] = v;
}

// c_in = [node_mem(172) | ao_o(100)] -> f16 (2B,288)
__global__ void pack_cin(const _Float16* __restrict__ node16, const _Float16* __restrict__ aoO,
                         _Float16* __restrict__ cin) {
  int b = blockIdx.x;
  for (int c = threadIdx.x; c < 288; c += blockDim.x) {
    _Float16 v;
    if (c < 172)      v = node16[(size_t)b * 192 + c];
    else if (c < 272) v = aoO[(size_t)b * 128 + (c - 172)];
    else              v = (_Float16)0.f;
    cin[(size_t)b * 288 + c] = v;
  }
}

// p_in = [z_src(100) | z_dst(100)] -> f16 (B,224)
__global__ void pack_pin(const _Float16* __restrict__ z, _Float16* __restrict__ pin) {
  int i = blockIdx.x;
  for (int c = threadIdx.x; c < 224; c += blockDim.x) {
    _Float16 v = (_Float16)0.f;
    if (c < 100)      v = z[(size_t)i * 128 + c];
    else if (c < 200) v = z[(size_t)(BSZ + i) * 128 + (c - 100)];
    pin[(size_t)i * 224 + c] = v;
  }
}

// score = sigmoid(h1 @ p_W2^T + p_b2), N=1 -> plain VALU dot
__global__ void score_kernel(const _Float16* __restrict__ h1, const float* __restrict__ pW2,
                             const float* __restrict__ pb2, float* __restrict__ out) {
  int i = blockIdx.x * blockDim.x + threadIdx.x;
  if (i >= BSZ) return;
  float s = pb2[0];
  for (int d = 0; d < 128; ++d) s += (float)h1[(size_t)i * 128 + d] * pW2[d];
  out[i] = 1.f / (1.f + __expf(-s));
}

// ---------------------------------------------------------------------------
extern "C" void kernel_launch(void* const* d_in, const int* in_sizes, int n_in,
                              void* d_out, int out_size, void* d_ws, size_t ws_size,
                              hipStream_t stream) {
  (void)in_sizes; (void)n_in; (void)out_size; (void)ws_size;
  const float* memory  = (const float*)d_in[0];
  const int*   src_ids = (const int*)d_in[1];
  const int*   dst_ids = (const int*)d_in[2];
  const float* dt      = (const float*)d_in[3];
  const int*   n_ids   = (const int*)d_in[4];
  const float* n_dt    = (const float*)d_in[5];
  const float* t_w = (const float*)d_in[6];
  const float* t_b = (const float*)d_in[7];
  const float* m_W1 = (const float*)d_in[8];
  const float* m_b1 = (const float*)d_in[9];
  const float* m_W2 = (const float*)d_in[10];
  const float* m_b2 = (const float*)d_in[11];
  const float* g_Wih = (const float*)d_in[12];
  const float* g_Whh = (const float*)d_in[13];
  const float* g_bih = (const float*)d_in[14];
  const float* g_bhh = (const float*)d_in[15];
  const float* k_W = (const float*)d_in[16];
  const float* k_b = (const float*)d_in[17];
  const float* v_W = (const float*)d_in[18];
  const float* v_b = (const float*)d_in[19];
  const float* q_W = (const float*)d_in[20];
  const float* q_b = (const float*)d_in[21];
  const float* in_W = (const float*)d_in[22];
  const float* in_b = (const float*)d_in[23];
  const float* o_W = (const float*)d_in[24];
  const float* o_b = (const float*)d_in[25];
  const float* c_W = (const float*)d_in[26];
  const float* c_b = (const float*)d_in[27];
  const float* p_W1 = (const float*)d_in[28];
  const float* p_b1 = (const float*)d_in[29];
  const float* p_W2 = (const float*)d_in[30];
  const float* p_b2 = (const float*)d_in[31];

  char* wp = (char*)d_ws;
  auto alloc = [&](size_t bytes) -> void* {
    void* r = (void*)wp;
    wp += (bytes + 255) & ~(size_t)255;
    return r;
  };
  const size_t B = BSZ, B2 = 2 * BSZ, NB = NBROW;

  _Float16* x16    = (_Float16*)alloc(B  * 448 * 2);
  _Float16* ssrc16 = (_Float16*)alloc(B  * 192 * 2);
  _Float16* sdst16 = (_Float16*)alloc(B  * 192 * 2);
  _Float16* hm16   = (_Float16*)alloc(B  * 128 * 2);
  _Float16* raw16  = (_Float16*)alloc(B  * 128 * 2);
  float*    gi     = (float*)   alloc(B  * 516 * 4);
  float*    gh     = (float*)   alloc(B  * 516 * 4);
  _Float16* newm16 = (_Float16*)alloc(B2 * 192 * 2);
  int*      upd    = (int*)     alloc((size_t)NMEM * 4);
  _Float16* node16 = (_Float16*)alloc(B2 * 192 * 2);
  _Float16* nbr16  = (_Float16*)alloc(NB * 288 * 2);   // reused for kh/vh later
  _Float16* keys16 = (_Float16*)alloc(NB * 128 * 2);
  _Float16* vals16 = (_Float16*)alloc(NB * 128 * 2);
  _Float16* q16    = (_Float16*)alloc(B2 * 128 * 2);
  _Float16* qh16   = (_Float16*)alloc(B2 * 128 * 2);
  _Float16* ao16   = (_Float16*)alloc(B2 * 128 * 2);
  _Float16* aoO16  = (_Float16*)alloc(B2 * 128 * 2);
  _Float16* cin16  = (_Float16*)alloc(B2 * 288 * 2);
  _Float16* z16    = (_Float16*)alloc(B2 * 128 * 2);
  _Float16* pin16  = (_Float16*)alloc(B  * 224 * 2);
  _Float16* h116   = (_Float16*)alloc(B  * 128 * 2);
  // kh/vh alias the (dead) nbr16 staging region: NB*512B <= NB*576B
  _Float16* kh16 = nbr16;
  _Float16* vh16 = nbr16 + NB * 128;

  _Float16* w_mW1  = (_Float16*)alloc(128 * 448 * 2);
  _Float16* w_mW2  = (_Float16*)alloc(128 * 128 * 2);
  _Float16* w_gWih = (_Float16*)alloc(576 * 128 * 2);  // 36 col tiles (zero pad)
  _Float16* w_gWhh = (_Float16*)alloc(576 * 192 * 2);  // 36 col tiles (zero pad)
  _Float16* w_kW   = (_Float16*)alloc(128 * 288 * 2);
  _Float16* w_vW   = (_Float16*)alloc(128 * 288 * 2);
  _Float16* w_qW   = (_Float16*)alloc(128 * 192 * 2);
  _Float16* w_Wq   = (_Float16*)alloc(128 * 128 * 2);
  _Float16* w_Wk   = (_Float16*)alloc(128 * 128 * 2);
  _Float16* w_Wv   = (_Float16*)alloc(128 * 128 * 2);
  _Float16* w_oW   = (_Float16*)alloc(128 * 128 * 2);
  _Float16* w_cW   = (_Float16*)alloc(128 * 288 * 2);
  _Float16* w_pW1  = (_Float16*)alloc(128 * 224 * 2);

  // weight staging (f32 -> padded f16)
  cvt_weight<<<128, 128, 0, stream>>>(m_W1, 100, 444, 0, w_mW1, 448);
  cvt_weight<<<128, 128, 0, stream>>>(m_W2, 100, 100, 0, w_mW2, 128);
  cvt_weight<<<576, 128, 0, stream>>>(g_Wih, 516, 100, 0, w_gWih, 128);
  cvt_weight<<<576, 128, 0, stream>>>(g_Whh, 516, 172, 0, w_gWhh, 192);
  cvt_weight<<<128, 128, 0, stream>>>(k_W, 100, 272, 0, w_kW, 288);
  cvt_weight<<<128, 128, 0, stream>>>(v_W, 100, 272, 0, w_vW, 288);
  cvt_weight<<<128, 128, 0, stream>>>(q_W, 100, 172, 0, w_qW, 192);
  cvt_weight<<<128, 128, 0, stream>>>(in_W, 100, 100, 0,   w_Wq, 128);
  cvt_weight<<<128, 128, 0, stream>>>(in_W, 100, 100, 100, w_Wk, 128);
  cvt_weight<<<128, 128, 0, stream>>>(in_W, 100, 100, 200, w_Wv, 128);
  cvt_weight<<<128, 128, 0, stream>>>(o_W, 100, 100, 0, w_oW, 128);
  cvt_weight<<<128, 128, 0, stream>>>(c_W, 100, 272, 0, w_cW, 288);
  cvt_weight<<<128, 128, 0, stream>>>(p_W1, 128, 200, 0, w_pW1, 224);

  // grid.y * 4 tiles == total column tiles exactly (8 or 36)
#define GEMM(M, A, lda, W, ldw, bias, Nv, Nt, oF, oH, ldo, relu)                      \
  gemm_wmma_f16<4><<<dim3((M) / 16, (Nt) / 4), 32, 0, stream>>>(                      \
      A, lda, W, ldw, bias, Nv, oF, oH, ldo, relu)

  // message MLP
  build_x<<<(int)B, 128, 0, stream>>>(memory, src_ids, dst_ids, dt, t_w, t_b,
                                      x16, ssrc16, sdst16);
  GEMM(B, x16, 448, w_mW1, 448, m_b1, 100, 8, (float*)nullptr, hm16, 128, 1);
  GEMM(B, hm16, 128, w_mW2, 128, m_b2, 100, 8, (float*)nullptr, raw16, 128, 0);

  // GRU (gi shared; gh per endpoint, buffer reused)
  GEMM(B, raw16, 128, w_gWih, 128, g_bih, 516, 36, gi, (_Float16*)nullptr, 0, 0);
  GEMM(B, ssrc16, 192, w_gWhh, 192, g_bhh, 516, 36, gh, (_Float16*)nullptr, 0, 0);
  gru_kernel<<<(int)B, 192, 0, stream>>>(gi, gh, memory, src_ids, newm16, 0);
  GEMM(B, sdst16, 192, w_gWhh, 192, g_bhh, 516, 36, gh, (_Float16*)nullptr, 0, 0);
  gru_kernel<<<(int)B, 192, 0, stream>>>(gi, gh, memory, dst_ids, newm16, (int)B);

  // scatter index (dst wins), then post-update gathers
  fill_upd<<<(NMEM + 255) / 256, 256, 0, stream>>>(upd, NMEM);
  scatter_upd<<<((int)B + 255) / 256, 256, 0, stream>>>(upd, src_ids, 0, (int)B);
  scatter_upd<<<((int)B + 255) / 256, 256, 0, stream>>>(upd, dst_ids, (int)B, (int)B);
  gather_nodes<<<(int)B2, 128, 0, stream>>>(memory, newm16, upd, src_ids, dst_ids, node16);
  gather_nbrs<<<(int)NB, 128, 0, stream>>>(memory, newm16, upd, n_ids, n_dt, t_w, t_b, nbr16);

  // neighbor projections (dominant GEMMs) + query
  GEMM(NB, nbr16, 288, w_kW, 288, k_b, 100, 8, (float*)nullptr, keys16, 128, 0);
  GEMM(NB, nbr16, 288, w_vW, 288, v_b, 100, 8, (float*)nullptr, vals16, 128, 0);
  GEMM(B2, node16, 192, w_qW, 192, q_b, 100, 8, (float*)nullptr, q16, 128, 0);

  // in-projections (kh/vh alias nbr16 region — nbr16 is dead here)
  GEMM(B2, q16, 128, w_Wq, 128, in_b, 100, 8, (float*)nullptr, qh16, 128, 0);
  GEMM(NB, keys16, 128, w_Wk, 128, in_b + 100, 100, 8, (float*)nullptr, kh16, 128, 0);
  GEMM(NB, vals16, 128, w_Wv, 128, in_b + 200, 100, 8, (float*)nullptr, vh16, 128, 0);

  // attention + output projection
  attn_kernel<<<(int)B2, 128, 0, stream>>>(qh16, kh16, vh16, ao16);
  GEMM(B2, ao16, 128, w_oW, 128, o_b, 100, 8, (float*)nullptr, aoO16, 128, 0);

  // combine + predictor
  pack_cin<<<(int)B2, 128, 0, stream>>>(node16, aoO16, cin16);
  GEMM(B2, cin16, 288, w_cW, 288, c_b, 100, 8, (float*)nullptr, z16, 128, 1);
  pack_pin<<<(int)B, 128, 0, stream>>>(z16, pin16);
  GEMM(B, pin16, 224, w_pW1, 224, p_b1, 128, 8, (float*)nullptr, h116, 128, 1);
  score_kernel<<<((int)B + 255) / 256, 256, 0, stream>>>(h116, p_W2, p_b2, (float*)d_out);
#undef GEMM
}
